// MinimalAttention_80427557585580
// MI455X (gfx1250) — matrix operand order repaired and checked by reading, not compiled, not run on hardware
//
#include <hip/hip_runtime.h>
#include <hip/hip_bf16.h>

// MI455X / gfx1250, wave32. bf16 WMMA (16x16x32), f32 accumulate.
// Attention computed TRANSPOSED (S^T = K Q^T, O^T = V^T P^T) so softmax
// reductions are in-register + one cross-half shuffle per block.

typedef __attribute__((ext_vector_type(16))) __bf16 v16bf;
typedef __attribute__((ext_vector_type(8)))  __bf16 v8bf;
typedef __attribute__((ext_vector_type(2)))  __bf16 v2bf;
typedef __attribute__((ext_vector_type(8)))  float  v8f;

#define SEQ     4096
#define DMODEL  768
#define QKV_N   2304
#define NH      12
#define DH      64
#define LDQKV   2304

// ---- fp32 -> bf16 (RNE), packed pair -> one dword ---------------------------
__device__ __forceinline__ unsigned pack2bf(float lo, float hi) {
#if __has_builtin(__builtin_amdgcn_cvt_pk_bf16_f32)
  return __builtin_bit_cast(unsigned, __builtin_amdgcn_cvt_pk_bf16_f32(lo, hi));
#else
  unsigned a = __builtin_bit_cast(unsigned, lo);
  unsigned b = __builtin_bit_cast(unsigned, hi);
  a = (a + 0x7FFFu + ((a >> 16) & 1u)) >> 16;
  b = (b + 0x7FFFu + ((b >> 16) & 1u)) & 0xFFFF0000u;
  return a | b;
#endif
}
__device__ __forceinline__ __bf16 f2bf(float x) {
  unsigned p = pack2bf(x, x);
  return __builtin_bit_cast(v2bf, p)[0];
}
__device__ __forceinline__ v16bf cat8(v8bf lo, v8bf hi) {
  return __builtin_shufflevector(lo, hi, 0, 1, 2, 3, 4, 5, 6, 7,
                                 8, 9, 10, 11, 12, 13, 14, 15);
}
__device__ __forceinline__ v8bf ld8(const __bf16* p) { return *(const v8bf*)p; }

// A-fragment (16x32, MxK): lane m = lane%16, hlf = lane/16.
// element j: k = j + 8*hlf (j<8) ; k = 16 + 8*hlf + (j-8) -> two 16B runs.
__device__ __forceinline__ v16bf a_frag(const __bf16* row, int hlf) {
  return cat8(ld8(row + 8 * hlf), ld8(row + 16 + 8 * hlf));
}
// B-fragment (32x16, KxN): lane n = lane%16, hlf = lane/16.
// element j: k = j + 8*hlf (j<8) ; k = 16 + 8*hlf + (j-8) -> two 16B runs.
__device__ __forceinline__ v16bf b_frag(const __bf16* row, int hlf) {
  return cat8(ld8(row + 8 * hlf), ld8(row + 16 + 8 * hlf));
}

#define WMMA_BF16(A, B, C) \
  __builtin_amdgcn_wmma_f32_16x16x32_bf16(false, (A), false, (B), (short)0, (C), false, false)

// ---------------------------------------------------------------------------
// elementwise fp32 -> bf16 pairs, with uniform scale (grid-stride)
// ---------------------------------------------------------------------------
__global__ __launch_bounds__(256) void cvt_f32_bf16(
    const float* __restrict__ in, unsigned* __restrict__ out, int npairs, float scale) {
  int i = blockIdx.x * 256 + threadIdx.x;
  const int stride = gridDim.x * 256;
  for (; i < npairs; i += stride) {
    float2 f = ((const float2*)in)[i];
    out[i] = pack2bf(f.x * scale, f.y * scale);
  }
}

// scaled bias copy: first nscaled entries multiplied by scale
__global__ __launch_bounds__(256) void scale_bias(
    const float* __restrict__ in, float* __restrict__ out, int n, int nscaled, float scale) {
  int i = blockIdx.x * 256 + threadIdx.x;
  if (i < n) out[i] = in[i] * (i < nscaled ? scale : 1.0f);
}

// ---------------------------------------------------------------------------
// C[M,N] = A[M,K] @ W[N,K]^T + bias[N]  (torch Linear layout), A/W bf16.
// 256 threads = 8 waves; wave tile 32x64 (8 WMMA / 32-k step); block 256x64.
// ---------------------------------------------------------------------------
template <bool OUT_BF16>
__global__ __launch_bounds__(256) void gemm_bias_bf16(
    const __bf16* __restrict__ A, const __bf16* __restrict__ W,
    const float* __restrict__ bias, void* __restrict__ Cout,
    int M, int N, int K) {
  const int lane = threadIdx.x & 31;
  const int wave = threadIdx.x >> 5;
  const int hlf  = lane >> 4;
  const int lidx = lane & 15;
  const int m0 = blockIdx.x * 256 + wave * 32;
  const int n0 = blockIdx.y * 64;

  v8f acc0[4] = {}, acc1[4] = {};
  const __bf16* arow0 = A + (size_t)(m0 + lidx) * K;
  const __bf16* arow1 = A + (size_t)(m0 + 16 + lidx) * K;

  for (int k0 = 0; k0 < K; k0 += 32) {
    v16bf a0 = a_frag(arow0 + k0, hlf);
    v16bf a1 = a_frag(arow1 + k0, hlf);
#pragma unroll
    for (int t = 0; t < 4; ++t) {
      const __bf16* wrow = W + (size_t)(n0 + t * 16 + lidx) * K + k0;
      v16bf b = b_frag(wrow, hlf);
      acc0[t] = WMMA_BF16(a0, b, acc0[t]);
      acc1[t] = WMMA_BF16(a1, b, acc1[t]);
    }
  }
#pragma unroll
  for (int t = 0; t < 4; ++t) {
    const int col = n0 + t * 16 + lidx;
    const float bv = bias[col];
#pragma unroll
    for (int r = 0; r < 8; ++r) {
      const size_t row0 = (size_t)(m0 + hlf * 8 + r);
      const float v0 = acc0[t][r] + bv;
      const float v1 = acc1[t][r] + bv;
      if (OUT_BF16) {
        ((__bf16*)Cout)[row0 * N + col]        = f2bf(v0);
        ((__bf16*)Cout)[(row0 + 16) * N + col] = f2bf(v1);
      } else {
        ((float*)Cout)[row0 * N + col]        = v0;
        ((float*)Cout)[(row0 + 16) * N + col] = v1;
      }
    }
  }
}

// ---------------------------------------------------------------------------
// Fused attention, transposed. Q section of qkv is pre-scaled by 1/sqrt(DH).
// Reference semantics: softmax over ALL keys; causal mask AFTER normalization.
// 8 waves; wave = 16 queries; grid = (SEQ/128, NH). Key blocks of 32.
//   S^T tiles: keys in VGPRs, queries in lanes -> per-lane scalar m/l stats.
//   O^T accum: dh in VGPRs, queries in lanes  -> scalar corr rescale.
// ---------------------------------------------------------------------------
__global__ __launch_bounds__(256) void attn_fused(
    const __bf16* __restrict__ qkv, __bf16* __restrict__ y) {
  __shared__ __bf16 Vt[DH][32];       // transposed V tile: Vt[d][key]
  __shared__ __bf16 Pst[8][16 * 32];  // per-wave P^T staging: Pst[q][key]

  const int lane = threadIdx.x & 31;
  const int wave = threadIdx.x >> 5;
  const int hlf  = lane >> 4;
  const int lidx = lane & 15;
  const int h  = blockIdx.y;
  const int q0 = blockIdx.x * 128 + wave * 16;

  const __bf16* Qp = qkv + h * DH;
  const __bf16* Kp = qkv + DMODEL + h * DH;
  const __bf16* Vp = qkv + 2 * DMODEL + h * DH;

  // Q as B-fragments (feat x query): lane n = query, contiguous feat runs
  v16bf qb0, qb1;
  {
    const __bf16* qrow = Qp + (size_t)(q0 + lidx) * LDQKV;
    qb0 = b_frag(qrow, hlf);        // features [0,32)
    qb1 = b_frag(qrow + 32, hlf);   // features [32,64)
  }

  v8f acc[4] = {};                  // O^T: acc[t4][r] = dh (t4*16+8*hlf+r), query lidx
  float mq = -__builtin_inff(), lq = 0.0f;
  const int qg = q0 + lidx;         // this lane's global query index

  __bf16* myP = &Pst[wave][0];
  const int tkey = threadIdx.x >> 3;        // 0..31 staging key
  const int td0  = (threadIdx.x & 7) * 8;   // staging feature base

  for (int kb = 0; kb < SEQ / 32; ++kb) {
    // ---- cooperative V staging (transpose into LDS), shared by all 8 waves
    __syncthreads();
    {
      v8bf g = ld8(Vp + (size_t)(kb * 32 + tkey) * LDQKV + td0);
#pragma unroll
      for (int j = 0; j < 8; ++j) Vt[td0 + j][tkey] = g[j];
    }

    // ---- S^T = K_blk(32x64) @ Q^T(64x16): two key tiles, 2 feature steps
    v8f st0 = {}, st1 = {};
    {
      const __bf16* ka0 = Kp + (size_t)(kb * 32 + lidx) * LDQKV;
      st0 = WMMA_BF16(a_frag(ka0, hlf), qb0, st0);
      st0 = WMMA_BF16(a_frag(ka0 + 32, hlf), qb1, st0);
      const __bf16* ka1 = Kp + (size_t)(kb * 32 + 16 + lidx) * LDQKV;
      st1 = WMMA_BF16(a_frag(ka1, hlf), qb0, st1);
      st1 = WMMA_BF16(a_frag(ka1 + 32, hlf), qb1, st1);
    }

    // ---- per-query stats: in-register tree + ONE cross-half shuffle
    float t0 = fmaxf(fmaxf(fmaxf(st0[0], st0[1]), fmaxf(st0[2], st0[3])),
                     fmaxf(fmaxf(st0[4], st0[5]), fmaxf(st0[6], st0[7])));
    float t1 = fmaxf(fmaxf(fmaxf(st1[0], st1[1]), fmaxf(st1[2], st1[3])),
                     fmaxf(fmaxf(st1[4], st1[5]), fmaxf(st1[6], st1[7])));
    float tmax = fmaxf(t0, t1);
    tmax = fmaxf(tmax, __shfl_xor(tmax, 16, 32));
    const float mnew = fmaxf(mq, tmax);
    const float corr = __expf(mq - mnew);
    mq = mnew;
#pragma unroll
    for (int t = 0; t < 4; ++t)
#pragma unroll
      for (int r = 0; r < 8; ++r) acc[t][r] *= corr;  // ALWAYS (future keys raise max)

    float e0[8], e1[8];
#pragma unroll
    for (int r = 0; r < 8; ++r) {
      e0[r] = __expf(st0[r] - mnew);
      e1[r] = __expf(st1[r] - mnew);
    }
    float psum = (((e0[0] + e0[1]) + (e0[2] + e0[3])) + ((e0[4] + e0[5]) + (e0[6] + e0[7]))) +
                 (((e1[0] + e1[1]) + (e1[2] + e1[3])) + ((e1[4] + e1[5]) + (e1[6] + e1[7])));
    psum += __shfl_xor(psum, 16, 32);
    lq = lq * corr + psum;          // denominator: ALL keys (unmasked)

    const bool active = (kb * 32) <= (q0 + 15);  // wave-uniform
    if (active) {
      // post-softmax causal mask (numerator only); pack 8 -> b128 LDS stores
      const int kb0 = kb * 32 + 8 * hlf;       // key of e0[r] = kb0 + r
      const int kb1 = kb * 32 + 16 + 8 * hlf;  // key of e1[r] = kb1 + r
      unsigned p[4];
#pragma unroll
      for (int i = 0; i < 4; ++i) {
        float a = (kb0 + 2 * i)     <= qg ? e0[2 * i]     : 0.0f;
        float b = (kb0 + 2 * i + 1) <= qg ? e0[2 * i + 1] : 0.0f;
        p[i] = pack2bf(a, b);
      }
      *(uint4*)(myP + lidx * 32 + 8 * hlf) = make_uint4(p[0], p[1], p[2], p[3]);
#pragma unroll
      for (int i = 0; i < 4; ++i) {
        float a = (kb1 + 2 * i)     <= qg ? e1[2 * i]     : 0.0f;
        float b = (kb1 + 2 * i + 1) <= qg ? e1[2 * i + 1] : 0.0f;
        p[i] = pack2bf(a, b);
      }
      *(uint4*)(myP + lidx * 32 + 16 + 8 * hlf) = make_uint4(p[0], p[1], p[2], p[3]);
    }

    __syncthreads();  // Vt staged; same-wave P stores ordered by DS pipeline

    if (active) {
      // O^T += V^T_tile(16x32) @ P^T(32x16): A from transposed Vt, B from Pst
      v16bf pb = b_frag(myP + lidx * 32, hlf);
      acc[0] = WMMA_BF16(a_frag(&Vt[lidx][0], hlf),      pb, acc[0]);
      acc[1] = WMMA_BF16(a_frag(&Vt[16 + lidx][0], hlf), pb, acc[1]);
      acc[2] = WMMA_BF16(a_frag(&Vt[32 + lidx][0], hlf), pb, acc[2]);
      acc[3] = WMMA_BF16(a_frag(&Vt[48 + lidx][0], hlf), pb, acc[3]);
    }
  }

  // ---- epilogue: one reciprocal per lane; packed b128 stores along dh
  const float inv = 1.0f / lq;
  __bf16* yrow = y + (size_t)qg * DMODEL + h * DH;
#pragma unroll
  for (int t4 = 0; t4 < 4; ++t4) {
    unsigned p0 = pack2bf(acc[t4][0] * inv, acc[t4][1] * inv);
    unsigned p1 = pack2bf(acc[t4][2] * inv, acc[t4][3] * inv);
    unsigned p2 = pack2bf(acc[t4][4] * inv, acc[t4][5] * inv);
    unsigned p3 = pack2bf(acc[t4][6] * inv, acc[t4][7] * inv);
    *(uint4*)(yrow + t4 * 16 + 8 * hlf) = make_uint4(p0, p1, p2, p3);
  }
}

// ---------------------------------------------------------------------------
// inputs: x[4096,768], W_qkv[2304,768], b_qkv[2304], W_out[768,768], b_out[768]
// out: [4096,768] f32.
// ws: xbf | wqkvb | woutb | qkvb | yattb (bf16) | sbias (f32)   ~36.2 MB
// ---------------------------------------------------------------------------
extern "C" void kernel_launch(void* const* d_in, const int* in_sizes, int n_in,
                              void* d_out, int out_size, void* d_ws, size_t ws_size,
                              hipStream_t stream) {
  const float* x     = (const float*)d_in[0];
  const float* W_qkv = (const float*)d_in[1];
  const float* b_qkv = (const float*)d_in[2];
  const float* W_out = (const float*)d_in[3];
  const float* b_out = (const float*)d_in[4];
  float* out = (float*)d_out;

  __bf16* xbf   = (__bf16*)d_ws;
  __bf16* wqkvb = xbf   + (size_t)SEQ * DMODEL;
  __bf16* woutb = wqkvb + (size_t)QKV_N * DMODEL;
  __bf16* qkvb  = woutb + (size_t)DMODEL * DMODEL;
  __bf16* yattb = qkvb  + (size_t)SEQ * QKV_N;
  float*  sbias = (float*)(yattb + (size_t)SEQ * DMODEL);

  const float qscale = 0.125f;  // 1/sqrt(DH), exact power of two

  // 0) pre-convert to bf16; fold qscale into the Q rows of W_qkv / b_qkv
  {
    const int np_x  = SEQ * DMODEL / 2;
    const int np_q  = DMODEL * DMODEL / 2;              // W_qkv rows [0,768)
    const int nf_q  = DMODEL * DMODEL;                  // floats in Q section
    const int np_kv = (QKV_N - DMODEL) * DMODEL / 2;    // W_qkv rows [768,2304)
    const int np_wo = DMODEL * DMODEL / 2;
    cvt_f32_bf16<<<(np_x + 255) / 256, 256, 0, stream>>>(x, (unsigned*)xbf, np_x, 1.0f);
    cvt_f32_bf16<<<(np_q + 255) / 256, 256, 0, stream>>>(W_qkv, (unsigned*)wqkvb, np_q, qscale);
    cvt_f32_bf16<<<(np_kv + 255) / 256, 256, 0, stream>>>(W_qkv + nf_q, (unsigned*)(wqkvb + nf_q), np_kv, 1.0f);
    cvt_f32_bf16<<<(np_wo + 255) / 256, 256, 0, stream>>>(W_out, (unsigned*)woutb, np_wo, 1.0f);
    scale_bias<<<(QKV_N + 255) / 256, 256, 0, stream>>>(b_qkv, sbias, QKV_N, DMODEL, qscale);
  }

  // 1) qkv = x @ W_qkv^T + b_qkv  (Q section pre-scaled; bf16 out)
  gemm_bias_bf16<true><<<dim3(SEQ / 256, QKV_N / 64), 256, 0, stream>>>(
      xbf, wqkvb, sbias, qkvb, SEQ, QKV_N, DMODEL);

  // 2) fused attention (full-row softmax, post-softmax causal mask), bf16 out
  attn_fused<<<dim3(SEQ / 128, NH), 256, 0, stream>>>(qkvb, yattb);

  // 3) out = y @ W_out^T + b_out  (f32 out)
  gemm_bias_bf16<false><<<dim3(SEQ / 256, DMODEL / 64), 256, 0, stream>>>(
      yattb, woutb, b_out, out, SEQ, DMODEL, DMODEL);
}
